// FastNLPPoetryModel_7181185319296
// MI455X (gfx1250) — compile-verified, matrix operand order checked by reading, not run on hardware
//
#include <hip/hip_runtime.h>
#include <hip/hip_bf16.h>

// Problem constants (fixed by setup_inputs)
#define VSZ 10000
#define ESZ 512
#define HSZ 1024
#define BSZ 64
#define SSZ 128
#define MROWS (SSZ * BSZ)      // 8192
#define G4H  (4 * HSZ)         // 4096

typedef __attribute__((ext_vector_type(16))) __bf16 v16bf;
typedef __attribute__((ext_vector_type(8)))  __bf16 v8bf;
typedef __attribute__((ext_vector_type(8)))  float  v8f;

union FragBF { v16bf v; v8bf h[2]; };

__device__ __forceinline__ __bf16 f2bf(float f) {
    union { float f; unsigned u; } in; in.f = f;
    unsigned u = in.u;
    u += 0x7FFFu + ((u >> 16) & 1u);          // round-to-nearest-even
    unsigned short s = (unsigned short)(u >> 16);
    union { unsigned short s; __bf16 b; } out; out.s = s;
    return out.b;
}

__device__ __forceinline__ float sigmoid_f(float x) {
    return 1.0f / (1.0f + __expf(-x));
}
__device__ __forceinline__ float tanh_f(float x) {
    x = fminf(fmaxf(x, -15.0f), 15.0f);
    float e = __expf(2.0f * x);
    return (e - 1.0f) / (e + 1.0f);
}

// Load one 16x32 bf16 A/B fragment. rowptr = this lane's row base (per-lane).
// ISA layout: lanes 0-15 hold K=kb..kb+7 (elems 0-7) and K=16+kb.. (elems 8-15),
// kb = (lane>=16)?8:0 -> two contiguous 16-byte loads per lane.
__device__ __forceinline__ v16bf load_frag_row(const __bf16* rowptr, int k) {
    int lane = threadIdx.x & 31;
    int kb = (lane & 16) ? 8 : 0;
    const __bf16* p = rowptr + k + kb;
    FragBF f;
    f.h[0] = *(const v8bf*)(p);
    f.h[1] = *(const v8bf*)(p + 16);
    return f.v;
}

__device__ __forceinline__ v8f wmma_bf16(v16bf a, v16bf b, v8f c) {
    return __builtin_amdgcn_wmma_f32_16x16x32_bf16(
        false, a, false, b, (short)0, c, false, false);
}

// ---------------------------------------------------------------------------
// prologue kernels
// ---------------------------------------------------------------------------
__global__ void cvt_f32_bf16_kernel(const float* __restrict__ src,
                                    __bf16* __restrict__ dst, int n) {
    int i = blockIdx.x * blockDim.x + threadIdx.x;
    int stride = gridDim.x * blockDim.x;
    for (; i < n; i += stride) dst[i] = f2bf(src[i]);
}

// x_bf[(s*B+b)*E + e] = bf16(emb[ids[b*S+s]*E + e])
__global__ void gather_emb_kernel(const int* __restrict__ ids,
                                  const float* __restrict__ emb,
                                  __bf16* __restrict__ x) {
    int i = blockIdx.x * blockDim.x + threadIdx.x;
    int stride = gridDim.x * blockDim.x;
    const int n = MROWS * ESZ;
    for (; i < n; i += stride) {
        int e   = i & (ESZ - 1);
        int row = i >> 9;            // /ESZ
        int s   = row >> 6;          // /BSZ
        int b   = row & (BSZ - 1);
        int id  = ids[b * SSZ + s];
        x[i] = f2bf(emb[(size_t)id * ESZ + e]);
    }
}

// ---------------------------------------------------------------------------
// bf16 GEMM: C[M,N] = A[M,K] @ B[N,K]^T + bias
// Block = 256 threads = 8 waves arranged 2(M) x 4(N); block tile 128 x 256.
// Wave tile 64 x 64 = 4x4 WMMA accumulators: 16 WMMAs per 8 fragment loads
// per K-step (1.0 b128 per WMMA). REMAP folds the batch-major transpose of
// hs into the A addressing for the logits GEMM.
// ---------------------------------------------------------------------------
template <bool REMAP>
__global__ void gemm_bf16_kernel(const __bf16* __restrict__ A,
                                 const __bf16* __restrict__ B,
                                 const float* __restrict__ bias,
                                 float* __restrict__ C,
                                 int M, int N, int K, int ldc, int nColBlocks) {
    int blk      = blockIdx.x;
    int rowBlk   = blk / nColBlocks;
    int colBlk   = blk % nColBlocks;
    int wave     = threadIdx.x >> 5;
    int wm       = wave >> 2;                 // 0..1
    int wn       = wave & 3;                  // 0..3
    int rowBase  = rowBlk * 128 + wm * 64;
    int colBase  = colBlk * 256 + wn * 64;

    int lane = threadIdx.x & 31;
    int lr   = lane & 15;

    // per-lane A row pointers (4 row-frags)
    const __bf16* arow[4];
#pragma unroll
    for (int i = 0; i < 4; ++i) {
        int r = rowBase + i * 16 + lr;
        if (REMAP) {                           // out[b*S+s] = hs[s*B+b]
            int b = r >> 7;                    // /SSZ
            int s = r & (SSZ - 1);
            r = s * BSZ + b;
        }
        arow[i] = A + (size_t)r * K;
    }
    // per-lane B row pointers (4 col-frags), clamped for N not mult of 16
    const __bf16* brow[4];
#pragma unroll
    for (int j = 0; j < 4; ++j) {
        int n = colBase + j * 16 + lr;
        if (n > N - 1) n = N - 1;
        brow[j] = B + (size_t)n * K;
    }

    v8f acc[4][4];
#pragma unroll
    for (int i = 0; i < 4; ++i)
#pragma unroll
        for (int j = 0; j < 4; ++j) acc[i][j] = (v8f){};

    for (int k = 0; k < K; k += 32) {
        v16bf a[4];
#pragma unroll
        for (int i = 0; i < 4; ++i) a[i] = load_frag_row(arow[i], k);
#pragma unroll
        for (int j = 0; j < 4; ++j) {
            v16bf bf = load_frag_row(brow[j], k);
#pragma unroll
            for (int i = 0; i < 4; ++i)
                acc[i][j] = wmma_bf16(a[i], bf, acc[i][j]);
        }
    }

    int colLane = lane & 15;
    int rb      = (lane >> 4) * 8;
    float bj[4];
#pragma unroll
    for (int j = 0; j < 4; ++j) {
        int cc = colBase + j * 16 + colLane;
        bj[j] = (cc < N) ? bias[cc] : 0.0f;
    }
#pragma unroll
    for (int i = 0; i < 4; ++i) {
#pragma unroll
        for (int j = 0; j < 4; ++j) {
            int cc = colBase + j * 16 + colLane;
            if (cc < N) {
#pragma unroll
                for (int v = 0; v < 8; ++v) {
                    int r = rowBase + i * 16 + rb + v;
                    C[(size_t)r * ldc + cc] = acc[i][j][v] + bj[j];
                }
            }
        }
    }
}

// ---------------------------------------------------------------------------
// One LSTM timestep, fused: g = xg_t + h_in @ W_hh^T; gates; c,h update.
// One wave computes a 32x16 (batch x hidden) tile for ALL four gates:
// per K-step, 2 A-frags are shared across 4 gate B-frags -> 8 WMMAs per
// 6 fragment loads. Grid: 16 blocks x 256 threads = 128 waves =
// (64/32 row tiles) * (1024/16 col tiles).
// ---------------------------------------------------------------------------
__global__ void lstm_step_kernel(const __bf16* __restrict__ h_in,
                                 const __bf16* __restrict__ Whh,
                                 const float* __restrict__ xg_t,
                                 float* __restrict__ c,
                                 float* __restrict__ h_f32,
                                 __bf16* __restrict__ h_out) {
    int wave = blockIdx.x * (blockDim.x >> 5) + (threadIdx.x >> 5);
    int mt = wave >> 6;                        // 0..1   batch tile (32 rows)
    int nt = wave & 63;                        // 0..63  hidden tile (16 cols)
    int row0 = mt * 32;
    int col0 = nt * 16;

    int lane = threadIdx.x & 31;
    int lr   = lane & 15;

    const __bf16* arow0 = h_in + (size_t)(row0 + lr) * HSZ;
    const __bf16* arow1 = h_in + (size_t)(row0 + 16 + lr) * HSZ;
    const __bf16* brg_[4];
#pragma unroll
    for (int g = 0; g < 4; ++g)
        brg_[g] = Whh + (size_t)(g * HSZ + col0 + lr) * HSZ;

    v8f acc[2][4];
#pragma unroll
    for (int i = 0; i < 2; ++i)
#pragma unroll
        for (int g = 0; g < 4; ++g) acc[i][g] = (v8f){};

    for (int k = 0; k < HSZ; k += 32) {
        v16bf a0 = load_frag_row(arow0, k);
        v16bf a1 = load_frag_row(arow1, k);
#pragma unroll
        for (int g = 0; g < 4; ++g) {
            v16bf bf = load_frag_row(brg_[g], k);
            acc[0][g] = wmma_bf16(a0, bf, acc[0][g]);
            acc[1][g] = wmma_bf16(a1, bf, acc[1][g]);
        }
    }

    int colLane = lane & 15;
    int rb      = (lane >> 4) * 8;
#pragma unroll
    for (int i = 0; i < 2; ++i) {
#pragma unroll
        for (int v = 0; v < 8; ++v) {
            int r  = row0 + i * 16 + rb + v;   // batch index
            int cc = col0 + colLane;           // hidden index
            const float* xr = xg_t + (size_t)r * G4H;
            float gi = acc[i][0][v] + xr[cc];
            float gf = acc[i][1][v] + xr[HSZ + cc];
            float gg = acc[i][2][v] + xr[2 * HSZ + cc];
            float go = acc[i][3][v] + xr[3 * HSZ + cc];
            float iS = sigmoid_f(gi);
            float fS = sigmoid_f(gf);
            float gT = tanh_f(gg);
            float oS = sigmoid_f(go);
            size_t idx = (size_t)r * HSZ + cc;
            float cn = fS * c[idx] + iS * gT;
            float hn = oS * tanh_f(cn);
            c[idx]     = cn;
            h_f32[idx] = hn;
            h_out[idx] = f2bf(hn);
        }
    }
}

// copy h_last / c_last (f32) into the tail of d_out
__global__ void copy_tail_kernel(const float* __restrict__ h,
                                 const float* __restrict__ c,
                                 float* __restrict__ out) {
    int i = blockIdx.x * blockDim.x + threadIdx.x;
    const size_t base = (size_t)MROWS * VSZ;
    if (i < BSZ * HSZ) {
        out[base + i]             = h[i];
        out[base + BSZ * HSZ + i] = c[i];
    }
}

// ---------------------------------------------------------------------------
extern "C" void kernel_launch(void* const* d_in, const int* in_sizes, int n_in,
                              void* d_out, int out_size, void* d_ws, size_t ws_size,
                              hipStream_t stream) {
    const int*   ids   = (const int*)  d_in[0];
    const float* emb   = (const float*)d_in[1];
    const float* W_ih  = (const float*)d_in[2];
    const float* W_hh  = (const float*)d_in[3];
    const float* bias  = (const float*)d_in[4];
    const float* W_out = (const float*)d_in[5];
    const float* b_out = (const float*)d_in[6];
    float* out = (float*)d_out;

    // workspace layout (256B aligned)
    char* ws = (char*)d_ws;
    size_t off = 0;
    auto take = [&](size_t bytes) {
        char* p = ws + off;
        off += (bytes + 255) & ~(size_t)255;
        return p;
    };
    __bf16* Wih_bf  = (__bf16*)take((size_t)G4H * ESZ * 2);       //  4 MB
    __bf16* Whh_bf  = (__bf16*)take((size_t)G4H * HSZ * 2);       //  8 MB
    __bf16* Wout_bf = (__bf16*)take((size_t)VSZ * HSZ * 2);       // 20 MB
    __bf16* x_bf    = (__bf16*)take((size_t)MROWS * ESZ * 2);     //  8 MB
    float*  xg      = (float*) take((size_t)MROWS * G4H * 4);     // 134 MB
    __bf16* hs_bf   = (__bf16*)take((size_t)SSZ * BSZ * HSZ * 2); // 17 MB
    __bf16* h0_bf   = (__bf16*)take((size_t)BSZ * HSZ * 2);
    float*  c_buf   = (float*) take((size_t)BSZ * HSZ * 4);
    float*  h_f32   = (float*) take((size_t)BSZ * HSZ * 4);
    (void)ws_size; (void)in_sizes; (void)n_in; (void)out_size;

    // 1) convert weights to bf16, gather+convert embeddings, zero state
    cvt_f32_bf16_kernel<<<2048, 256, 0, stream>>>(W_ih,  Wih_bf,  G4H * ESZ);
    cvt_f32_bf16_kernel<<<2048, 256, 0, stream>>>(W_hh,  Whh_bf,  G4H * HSZ);
    cvt_f32_bf16_kernel<<<2048, 256, 0, stream>>>(W_out, Wout_bf, VSZ * HSZ);
    gather_emb_kernel<<<2048, 256, 0, stream>>>(ids, emb, x_bf);
    hipMemsetAsync(h0_bf, 0, (size_t)BSZ * HSZ * 2, stream);
    hipMemsetAsync(c_buf, 0, (size_t)BSZ * HSZ * 4, stream);

    // 2) xg = x @ W_ih^T + b   (M=8192, N=4096, K=512)
    {
        int nColBlocks = G4H / 256;                  // 16
        int nRowBlocks = MROWS / 128;                // 64
        gemm_bf16_kernel<false><<<nRowBlocks * nColBlocks, 256, 0, stream>>>(
            x_bf, Wih_bf, bias, xg, MROWS, G4H, ESZ, G4H, nColBlocks);
    }

    // 3) sequential LSTM recurrence (128 dependent steps)
    for (int t = 0; t < SSZ; ++t) {
        const __bf16* h_in = (t == 0) ? h0_bf : hs_bf + (size_t)(t - 1) * BSZ * HSZ;
        __bf16* h_out = hs_bf + (size_t)t * BSZ * HSZ;
        const float* xg_t = xg + (size_t)t * BSZ * G4H;
        lstm_step_kernel<<<16, 256, 0, stream>>>(h_in, Whh_bf, xg_t,
                                                 c_buf, h_f32, h_out);
    }

    // 4) logits = reorder(hs) @ W_out^T + b_out  (M=8192, N=10000, K=1024)
    {
        int nColBlocks = (VSZ + 255) / 256;          // 40
        int nRowBlocks = MROWS / 128;                // 64
        gemm_bf16_kernel<true><<<nRowBlocks * nColBlocks, 256, 0, stream>>>(
            hs_bf, Wout_bf, b_out, out, MROWS, VSZ, HSZ, VSZ, nColBlocks);
    }

    // 5) h_last, c_last tail
    copy_tail_kernel<<<(BSZ * HSZ + 255) / 256, 256, 0, stream>>>(h_f32, c_buf, out);
}
